// MixtureOfExperts_9869834846615
// MI455X (gfx1250) — matrix operand order repaired and checked by reading, not compiled, not run on hardware
//
#include <hip/hip_runtime.h>
#include <hip/hip_bf16.h>

// -----------------------------------------------------------------------------
// MoE dense forward for MI455X (gfx1250), wave32.
//   gate = softmax(x@gW + gb)
//   h1 = relu(x@W0[e]+b0); h2 = relu(h1@W1[e]+b1); out += gate[:,e]*(h2@Wf[e]+bf)
// Compute-bound (~1.1 TFLOP vs ~300MB): everything runs through
// v_wmma_f32_16x16x32_bf16. Weights are transposed+converted to bf16 [N][K]
// per expert so ALL GEMM tiles are contiguous 16B chunks along K, staged into
// LDS with gfx1250 async-to-LDS copies (ASYNCcnt) and double buffering.
// -----------------------------------------------------------------------------

#define E_N   8
#define D_IN  1024
#define D_HID 2048
#define D_OUT 1024
#define N_TOK 8192

typedef __attribute__((ext_vector_type(16))) __bf16 v16bf;
typedef __attribute__((ext_vector_type(8)))  __bf16 v8bf;
typedef __attribute__((ext_vector_type(4)))  __bf16 v4bf;
typedef __attribute__((ext_vector_type(8)))  float  v8f;
typedef int v4i_vs __attribute__((__vector_size__(4 * sizeof(int))));

#if __has_builtin(__builtin_amdgcn_global_load_async_to_lds_b128)
#define HAVE_ASYNC_LDS 1
#else
#define HAVE_ASYNC_LDS 0
#endif

// 16B global -> LDS copy (per lane). Async path avoids the VGPR round-trip.
__device__ __forceinline__ void cp16_g2l(const __bf16* g, __bf16* l) {
#if HAVE_ASYNC_LDS
  __builtin_amdgcn_global_load_async_to_lds_b128(
      (__attribute__((address_space(1))) v4i_vs*)(g),
      (__attribute__((address_space(3))) v4i_vs*)(l), 0, 0);
#else
  *reinterpret_cast<v8bf*>(l) = *reinterpret_cast<const v8bf*>(g);
#endif
}

template <int N>
__device__ __forceinline__ void wait_async() {
#if HAVE_ASYNC_LDS
#if __has_builtin(__builtin_amdgcn_s_wait_asynccnt)
  __builtin_amdgcn_s_wait_asynccnt(N);
#else
  asm volatile("s_wait_asynccnt %0" ::"i"(N) : "memory");
#endif
#endif
}

__device__ __forceinline__ v16bf frag16(const __bf16* p) {
  const v8bf lo = *reinterpret_cast<const v8bf*>(p);
  const v8bf hi = *reinterpret_cast<const v8bf*>(p + 8);
  return __builtin_shufflevector(lo, hi, 0, 1, 2, 3, 4, 5, 6, 7,
                                 8, 9, 10, 11, 12, 13, 14, 15);
}

// ---------------- fp32 -> bf16 bulk convert (x) ------------------------------
__global__ void cvt_f32_to_bf16_v4(const float* __restrict__ src,
                                   __bf16* __restrict__ dst, int n4) {
  int i = blockIdx.x * blockDim.x + threadIdx.x;
  if (i >= n4) return;
  const float4 v = reinterpret_cast<const float4*>(src)[i];
  v4bf o;
  o.x = (__bf16)v.x; o.y = (__bf16)v.y; o.z = (__bf16)v.z; o.w = (__bf16)v.w;
  reinterpret_cast<v4bf*>(dst)[i] = o;
}

// ---------------- W[K][N] fp32 -> Wt[N][K] bf16 (tiled transpose) ------------
__global__ __launch_bounds__(256)
void transpose_cvt(const float* __restrict__ W, __bf16* __restrict__ Wt,
                   int K, int N) {
  __shared__ float tile[32][33];
  const int tx  = threadIdx.x & 31;
  const int ty0 = threadIdx.x >> 5;  // 0..7
  const int k0 = blockIdx.y * 32;
  const int n0 = blockIdx.x * 32;
#pragma unroll
  for (int i = 0; i < 4; ++i) {
    const int ty = ty0 + i * 8;
    tile[ty][tx] = W[(size_t)(k0 + ty) * N + n0 + tx];
  }
  __syncthreads();
#pragma unroll
  for (int i = 0; i < 4; ++i) {
    const int ty = ty0 + i * 8;
    Wt[(size_t)(n0 + ty) * K + k0 + tx] = (__bf16)tile[tx][ty];
  }
}

// ---------------- gate: logits + softmax, one wave32 per token ---------------
__global__ __launch_bounds__(256)
void gate_kernel(const float* __restrict__ x, const float* __restrict__ gW,
                 const float* __restrict__ gb, float* __restrict__ gate) {
  const int lane = threadIdx.x & 31;
  const int wave = threadIdx.x >> 5;
  const int row  = blockIdx.x * 8 + wave;
  float acc[E_N];
#pragma unroll
  for (int e = 0; e < E_N; ++e) acc[e] = 0.0f;
  const float* xr = x + (size_t)row * D_IN;
  for (int d = lane; d < D_IN; d += 32) {
    const float xv = xr[d];
    const float* w = gW + (size_t)d * E_N;
#pragma unroll
    for (int e = 0; e < E_N; ++e) acc[e] += xv * w[e];
  }
#pragma unroll
  for (int e = 0; e < E_N; ++e) {
#pragma unroll
    for (int off = 16; off > 0; off >>= 1)
      acc[e] += __shfl_down(acc[e], off, 32);
  }
  if (lane == 0) {
    float mx = -1e30f;
#pragma unroll
    for (int e = 0; e < E_N; ++e) { acc[e] += gb[e]; mx = fmaxf(mx, acc[e]); }
    float s = 0.0f;
#pragma unroll
    for (int e = 0; e < E_N; ++e) { acc[e] = __expf(acc[e] - mx); s += acc[e]; }
    const float inv = 1.0f / s;
#pragma unroll
    for (int e = 0; e < E_N; ++e) gate[(size_t)row * E_N + e] = acc[e] * inv;
  }
}

// ---------------- tiled bf16 WMMA GEMM ---------------------------------------
// C[M,N] = A[M,K](bf16, row-major) * Bt[N,K](bf16, N-major) + bias
// MODE 0: ReLU, store bf16. MODE 1: scale by gate[row*8+e], f32 accum into out.
// BM=128, BN=128, BK=64; 256 threads = 8 waves, wave owns 64x32 patch.
// Double-buffered LDS, async global->LDS (8 x b128 per thread per tile).
template <int MODE>
__global__ __launch_bounds__(256)
void moe_wmma_gemm(const __bf16* __restrict__ A, int lda,
                   const __bf16* __restrict__ Bt, int ldb,
                   const float*  __restrict__ bias,
                   void* __restrict__ Cptr, int ldc,
                   const float* __restrict__ gate, int expert, int init,
                   int K) {
  constexpr int BK   = 64;
  constexpr int LDSB = 88;  // row stride (elems): 176B, 16B-aligned, bank-spread
  __shared__ __align__(16) __bf16 As[2][128 * LDSB];
  __shared__ __align__(16) __bf16 Bs[2][128 * LDSB];

  const int tid  = threadIdx.x;
  const int lane = tid & 31;
  const int wave = tid >> 5;
  const int wm = wave >> 2;          // 0..1
  const int wn = wave & 3;           // 0..3
  const int rr = lane & 15;
  const int hh = (lane >> 4) * 16;   // fragment half select

  const int m0 = blockIdx.y * 128;
  const int n0 = blockIdx.x * 128;

  // Per-thread tile-fill map: 4 chunks of 8 bf16 for A, 4 for B.
  // A rows natural; B rows natural.
  uint32_t a_lds[4], a_g[4], b_lds[4], b_g[4];
#pragma unroll
  for (int it = 0; it < 4; ++it) {
    const int c = tid + it * 256;       // 0..1023
    const int row = c >> 3, cc = c & 7; // 8 chunks per 64-wide row
    const int g = cc >> 2, s = cc & 3;
    const int p = ((s & 1) << 1) | (s >> 1);  // 0,2,1,3
    a_lds[it] = (uint32_t)(row * LDSB + cc * 8);
    b_lds[it] = (uint32_t)(row * LDSB + cc * 8);
    a_g[it] = (uint32_t)(m0 + row) * (uint32_t)lda + (uint32_t)(cc * 8);
    b_g[it] = (uint32_t)(n0 + row) * (uint32_t)ldb + (uint32_t)(cc * 8);
  }

  auto issue = [&](int kt, int buf) {
    const uint32_t ko = (uint32_t)kt * BK;
#pragma unroll
    for (int it = 0; it < 4; ++it)
      cp16_g2l(A + (size_t)(a_g[it] + ko), &As[buf][a_lds[it]]);
#pragma unroll
    for (int it = 0; it < 4; ++it)
      cp16_g2l(Bt + (size_t)(b_g[it] + ko), &Bs[buf][b_lds[it]]);
  };

  const v8f vz = {0.f, 0.f, 0.f, 0.f, 0.f, 0.f, 0.f, 0.f};
  v8f acc[4][2];
#pragma unroll
  for (int tm = 0; tm < 4; ++tm)
#pragma unroll
    for (int tn = 0; tn < 2; ++tn) acc[tm][tn] = vz;

  const int nk = K / BK;
  issue(0, 0);
  for (int kt = 0; kt < nk; ++kt) {
    const int cur = kt & 1;
    if (kt + 1 < nk) {
      issue(kt + 1, cur ^ 1);
      wait_async<8>();   // oldest 8 (tile kt) retired; in-order completion
    } else {
      wait_async<0>();
    }
    __syncthreads();     // tile kt visible to all waves

#pragma unroll
    for (int ks = 0; ks < 2; ++ks) {
      const int kb = ks * 32 + hh;
      v16bf bfrag[2];
#pragma unroll
      for (int tn = 0; tn < 2; ++tn)
        bfrag[tn] = frag16(&Bs[cur][(wn * 32 + tn * 16 + rr) * LDSB + kb]);
#pragma unroll
      for (int tm = 0; tm < 4; ++tm) {
        const v16bf afrag = frag16(&As[cur][(wm * 64 + tm * 16 + rr) * LDSB + kb]);
#pragma unroll
        for (int tn = 0; tn < 2; ++tn)
          acc[tm][tn] = __builtin_amdgcn_wmma_f32_16x16x32_bf16(
              false, afrag, false, bfrag[tn], (short)0, acc[tm][tn], false, false);
      }
    }
    __syncthreads();     // all waves done reading buf 'cur' before refill
  }

  // --- epilogue --------------------------------------------------------------
  if (MODE == 0) {
    __bf16* Co = reinterpret_cast<__bf16*>(Cptr);
#pragma unroll
    for (int tm = 0; tm < 4; ++tm) {
      const int rowbase = m0 + wm * 64 + tm * 16 + (lane >> 4) * 8;
#pragma unroll
      for (int tn = 0; tn < 2; ++tn) {
        const int col = n0 + wn * 32 + tn * 16 + rr;
        const float bc = bias[col];
#pragma unroll
        for (int i = 0; i < 8; ++i) {
          float v = acc[tm][tn][i] + bc;
          v = fmaxf(v, 0.0f);
          Co[(size_t)(rowbase + i) * ldc + col] = (__bf16)v;
        }
      }
    }
  } else {
    float* Co = reinterpret_cast<float*>(Cptr);
#pragma unroll
    for (int tm = 0; tm < 4; ++tm) {
      const int rowbase = m0 + wm * 64 + tm * 16 + (lane >> 4) * 8;
#pragma unroll
      for (int tn = 0; tn < 2; ++tn) {
        const int col = n0 + wn * 32 + tn * 16 + rr;
        const float bc = bias[col];
#pragma unroll
        for (int i = 0; i < 8; ++i) {
          const int row = rowbase + i;
          const float g = gate[(size_t)row * E_N + expert];
          const float v = (acc[tm][tn][i] + bc) * g;
          const size_t o = (size_t)row * ldc + col;
          if (init) Co[o] = v; else Co[o] += v;
        }
      }
    }
  }
}

// -----------------------------------------------------------------------------
extern "C" void kernel_launch(void* const* d_in, const int* in_sizes, int n_in,
                              void* d_out, int out_size, void* d_ws, size_t ws_size,
                              hipStream_t stream) {
  const float* x   = (const float*)d_in[0];
  const float* gW  = (const float*)d_in[1];
  const float* gb  = (const float*)d_in[2];
  const float* W0  = (const float*)d_in[3];
  const float* b0  = (const float*)d_in[4];
  const float* W1  = (const float*)d_in[5];
  const float* b1  = (const float*)d_in[6];
  const float* Wf  = (const float*)d_in[7];
  const float* bfv = (const float*)d_in[8];
  float* out = (float*)d_out;

  char* ws = (char*)d_ws;
  size_t off = 0;
  __bf16* x_bf = (__bf16*)(ws + off); off += (size_t)N_TOK * D_IN * 2;    // 16 MB
  float*  gate = (float*)(ws + off);  off += (size_t)N_TOK * E_N * 4;     // 256 KB
  __bf16* h1   = (__bf16*)(ws + off); off += (size_t)N_TOK * D_HID * 2;   // 32 MB
  __bf16* h2   = (__bf16*)(ws + off); off += (size_t)N_TOK * D_HID * 2;   // 32 MB
  __bf16* w0t  = (__bf16*)(ws + off); off += (size_t)D_HID * D_IN * 2;    //  4 MB
  __bf16* w1t  = (__bf16*)(ws + off); off += (size_t)D_HID * D_HID * 2;   //  8 MB
  __bf16* wft  = (__bf16*)(ws + off);                                     //  4 MB

  { // x -> bf16
    const int n4 = N_TOK * D_IN / 4;
    cvt_f32_to_bf16_v4<<<(n4 + 255) / 256, 256, 0, stream>>>(x, x_bf, n4);
  }
  gate_kernel<<<N_TOK / 8, 256, 0, stream>>>(x, gW, gb, gate);

  const dim3 blk(256);
  const dim3 gHid(D_HID / 128, N_TOK / 128);
  const dim3 gOut(D_OUT / 128, N_TOK / 128);
  for (int e = 0; e < E_N; ++e) {
    const float* W0e = W0 + (size_t)e * D_IN * D_HID;
    const float* W1e = W1 + (size_t)e * D_HID * D_HID;
    const float* Wfe = Wf + (size_t)e * D_HID * D_OUT;
    // weights -> bf16, transposed to [N][K] (stream-ordered before GEMMs)
    transpose_cvt<<<dim3(D_HID / 32, D_IN / 32), blk, 0, stream>>>(W0e, w0t, D_IN, D_HID);
    transpose_cvt<<<dim3(D_HID / 32, D_HID / 32), blk, 0, stream>>>(W1e, w1t, D_HID, D_HID);
    transpose_cvt<<<dim3(D_OUT / 32, D_HID / 32), blk, 0, stream>>>(Wfe, wft, D_HID, D_OUT);

    moe_wmma_gemm<0><<<gHid, blk, 0, stream>>>(
        x_bf, D_IN, w0t, D_IN, b0 + (size_t)e * D_HID,
        (void*)h1, D_HID, nullptr, 0, 0, D_IN);
    moe_wmma_gemm<0><<<gHid, blk, 0, stream>>>(
        h1, D_HID, w1t, D_HID, b1 + (size_t)e * D_HID,
        (void*)h2, D_HID, nullptr, 0, 0, D_HID);
    moe_wmma_gemm<1><<<gOut, blk, 0, stream>>>(
        h2, D_HID, wft, D_HID, bfv + (size_t)e * D_OUT,
        (void*)out, D_OUT, gate, e, (e == 0) ? 1 : 0, D_HID);
  }
}
